// ImMatchNet_31739808317592
// MI455X (gfx1250) — matrix operand order repaired and checked by reading, not compiled, not run on hardware
//
#include <hip/hip_runtime.h>

// ---------------------------------------------------------------------------
// ImMatchNet pipeline for MI455X (gfx1250, wave32, WMMA)
//   1) L2-norm features over C + convert to f16, laid out [b][p=h*48+w][c]
//   2) Correlation GEMM: f16 WMMA, f32 accum, LDS double-buffered with
//      async global->LDS copies (ASYNCcnt path), fused ReLU + x/sqrt(x^2+eps)
//   3) 4D max-pool k=2  (48^4 -> 24^4 per batch)
//   4) mutual matching (row/col max + elementwise)
//   5) 3x Conv4d(3^4, SAME) + ReLU
//   6) mutual matching -> d_out
// ---------------------------------------------------------------------------

#define EPS_L2 1e-6f
#define EPS_MM 1e-5f

typedef __attribute__((ext_vector_type(16))) _Float16 v16h;
typedef __attribute__((ext_vector_type(8)))  float    v8f;
typedef __attribute__((ext_vector_type(4)))  int      v4i;

union FragH { v16h v; uint4 q[2]; };

// --- gfx1250 async global->LDS path (guarded; falls back to VGPR staging) ---
#if defined(__has_builtin)
#if __has_builtin(__builtin_amdgcn_global_load_async_to_lds_b128) && \
    __has_builtin(__builtin_amdgcn_s_wait_asynccnt)
#define USE_ASYNC_LDS 1
#endif
#endif
#ifndef USE_ASYNC_LDS
#define USE_ASYNC_LDS 0
#endif

__device__ __forceinline__ void async_copy16(const _Float16* g, _Float16* l) {
#if USE_ASYNC_LDS
  __builtin_amdgcn_global_load_async_to_lds_b128(
      (__attribute__((address_space(1))) v4i*)(void*)g,
      (__attribute__((address_space(3))) v4i*)(void*)l, 0, 0);
#else
  *(uint4*)l = *(const uint4*)g;
#endif
}

__device__ __forceinline__ void stage_wait_barrier() {
#if USE_ASYNC_LDS
  __builtin_amdgcn_s_wait_asynccnt(0);
#endif
  __syncthreads();
}

// ---------------------------------------------------------------------------
// Kernel 1: channel L2-norm + f32->f16 + transpose [b][c][h][w] -> [b][p][c]
// block = (48 w, 8 c-partitions), grid = (48 h, 8 b)
// ---------------------------------------------------------------------------
__global__ __launch_bounds__(384) void l2norm_f16_kernel(
    const float* __restrict__ f, _Float16* __restrict__ out) {
  const int w  = threadIdx.x;   // 0..47
  const int cp = threadIdx.y;   // 0..7
  const int h  = blockIdx.x;
  const int b  = blockIdx.y;
  __shared__ float part[8][48];
  __shared__ float ninv[48];

  const size_t base = (size_t)b * 1024 * 2304;  // C*H*W = 1024*2304
  const size_t hw   = (size_t)h * 48 + w;

  float s = 0.f;
  for (int c = cp; c < 1024; c += 8) {
    float v = f[base + (size_t)c * 2304 + hw];
    s += v * v;
  }
  part[cp][w] = s;
  __syncthreads();
  if (cp == 0) {
    float t = 0.f;
#pragma unroll
    for (int i = 0; i < 8; ++i) t += part[i][w];
    ninv[w] = rsqrtf(t + EPS_L2);
  }
  __syncthreads();
  const float inv = ninv[w];
  _Float16* orow = out + ((size_t)b * 2304 + hw) * 1024;
  for (int c = cp; c < 1024; c += 8)
    orow[c] = (_Float16)(f[base + (size_t)c * 2304 + hw] * inv);
}

// ---------------------------------------------------------------------------
// Kernel 2: correlation GEMM via v_wmma_f32_16x16x32_f16
// corr[b][p][q] = sum_c faH[b][p][c] * fbH[b][q][c], fused ReLU + norm.
// Block 256 threads = 8 waves (4x2); block tile 128(p) x 128(q) staged in
// LDS (double buffered, async copies); wave tile 32x64 = 2x4 WMMA accums.
// LDS rows padded to 40 halves (80B): stride/4 = 20 -> 16 distinct banks
// across each half-wave for the ds_load_b128 fragment reads.
// grid = (18, 18, 8)
// ---------------------------------------------------------------------------
#define LDSP 40  // padded K pitch in halves

__global__ __launch_bounds__(256) void corr_wmma_kernel(
    const _Float16* __restrict__ faH, const _Float16* __restrict__ fbH,
    float* __restrict__ corr) {
  __shared__ _Float16 As[2][128 * LDSP];
  __shared__ _Float16 Bs[2][128 * LDSP];

  const int b    = blockIdx.z;
  const int tid  = threadIdx.x;
  const int wave = tid >> 5;
  const int lane = tid & 31;
  const int wr   = wave >> 1;       // 0..3 -> p offset wr*32
  const int wc   = wave & 1;        // 0..1 -> q offset wc*64
  const int lm   = lane & 15;
  const int hi   = lane >> 4;       // 0 or 1

  const _Float16* aTile = faH + ((size_t)b * 2304 + blockIdx.y * 128) * 1024;
  const _Float16* bTile = fbH + ((size_t)b * 2304 + blockIdx.x * 128) * 1024;

  // cooperative stage: 128 rows x 32 halves = 512 x 16B chunks per matrix
  const int r0 = tid >> 2, s0 = (tid & 3) * 8;          // chunk tid
  const int r1 = (tid + 256) >> 2, s1 = s0;             // chunk tid+256

  auto load_stage = [&](int buf, int kk) {
    async_copy16(aTile + (size_t)r0 * 1024 + kk + s0, &As[buf][r0 * LDSP + s0]);
    async_copy16(aTile + (size_t)r1 * 1024 + kk + s1, &As[buf][r1 * LDSP + s1]);
    async_copy16(bTile + (size_t)r0 * 1024 + kk + s0, &Bs[buf][r0 * LDSP + s0]);
    async_copy16(bTile + (size_t)r1 * 1024 + kk + s1, &Bs[buf][r1 * LDSP + s1]);
  };

  v8f acc[2][4];
#pragma unroll
  for (int mt = 0; mt < 2; ++mt)
#pragma unroll
    for (int nt = 0; nt < 4; ++nt) acc[mt][nt] = (v8f){};

  auto compute_stage = [&](int buf) {
    // A fragment (16x32 f16): lane = row M=lm, K chunks {off..off+7, off+16..off+23}
    const _Float16* aPtr = &As[buf][(wr * 32 + lm) * LDSP + hi * 8];
    // B fragment (32x16 f16): lane = col N=lm, K chunks {off..off+7, off+16..off+23}
    const _Float16* bPtr = &Bs[buf][(wc * 64 + lm) * LDSP + hi * 8];
    FragH a[2], bf[4];
#pragma unroll
    for (int mt = 0; mt < 2; ++mt) {
      const _Float16* p = aPtr + mt * 16 * LDSP;
      a[mt].q[0] = *(const uint4*)(p);
      a[mt].q[1] = *(const uint4*)(p + 16);
    }
#pragma unroll
    for (int nt = 0; nt < 4; ++nt) {
      const _Float16* p = bPtr + nt * 16 * LDSP;
      bf[nt].q[0] = *(const uint4*)(p);
      bf[nt].q[1] = *(const uint4*)(p + 16);
    }
#pragma unroll
    for (int mt = 0; mt < 2; ++mt)
#pragma unroll
      for (int nt = 0; nt < 4; ++nt)
        acc[mt][nt] = __builtin_amdgcn_wmma_f32_16x16x32_f16(
            false, a[mt].v, false, bf[nt].v, (short)0, acc[mt][nt], false, false);
  };

  load_stage(0, 0);
  stage_wait_barrier();
  for (int ks = 1; ks < 32; ++ks) {
    load_stage(ks & 1, ks * 32);      // prefetch next stage (overlaps WMMA)
    compute_stage((ks - 1) & 1);
    stage_wait_barrier();
  }
  compute_stage(1);  // ks=31 -> buf 1

  // Epilogue: D layout lane l, vgpr r -> M = r + 8*(l>>4), N = l&15
  // fused ReLU + x/sqrt(x^2+eps)
#pragma unroll
  for (int mt = 0; mt < 2; ++mt)
#pragma unroll
    for (int nt = 0; nt < 4; ++nt) {
      const int p = blockIdx.y * 128 + wr * 32 + mt * 16 + 8 * hi;
      const int q = blockIdx.x * 128 + wc * 64 + nt * 16 + lm;
      float* dst = corr + ((size_t)b * 2304 + p) * 2304 + q;
#pragma unroll
      for (int r = 0; r < 8; ++r) {
        float v = fmaxf(acc[mt][nt][r], 0.f);
        v = v * rsqrtf(v * v + EPS_L2);
        dst[(size_t)r * 2304] = v;
      }
    }
}

// ---------------------------------------------------------------------------
// Kernel 3: strided 4D max-pool k=2: [b][2304][2304] -> [b][576][576]
// ---------------------------------------------------------------------------
__global__ void pool4d_kernel(const float* __restrict__ corr,
                              float* __restrict__ pooled) {
  const int idx = blockIdx.x * blockDim.x + threadIdx.x;
  if (idx >= 8 * 576 * 576) return;
  const int qq = idx % 576;
  const int pp = (idx / 576) % 576;
  const int b  = idx / (576 * 576);
  const int ph = pp / 24, pw = pp % 24;
  const int qh = qq / 24, qw = qq % 24;
  float m = -1e30f;
#pragma unroll
  for (int dh = 0; dh < 2; ++dh)
#pragma unroll
    for (int dw = 0; dw < 2; ++dw) {
      const int p = (2 * ph + dh) * 48 + (2 * pw + dw);
      const float* row = corr + ((size_t)b * 2304 + p) * 2304;
#pragma unroll
      for (int eh = 0; eh < 2; ++eh)
#pragma unroll
        for (int ew = 0; ew < 2; ++ew) {
          const int q = (2 * qh + eh) * 48 + (2 * qw + ew);
          m = fmaxf(m, row[q]);
        }
    }
  pooled[idx] = m;
}

// ---------------------------------------------------------------------------
// Mutual matching helpers on [b][576][576]
// ---------------------------------------------------------------------------
__global__ void rowmax_kernel(const float* __restrict__ c,
                              float* __restrict__ amax) {
  __shared__ float red[64];
  const int row = blockIdx.x;              // b*576 + p
  const float* r = c + (size_t)row * 576;
  float m = -1e30f;
  for (int q = threadIdx.x; q < 576; q += 64) m = fmaxf(m, r[q]);
  red[threadIdx.x] = m;
  __syncthreads();
  for (int s = 32; s > 0; s >>= 1) {
    if (threadIdx.x < s) red[threadIdx.x] = fmaxf(red[threadIdx.x], red[threadIdx.x + s]);
    __syncthreads();
  }
  if (threadIdx.x == 0) amax[row] = red[0];
}

__global__ void colmax_kernel(const float* __restrict__ c,
                              float* __restrict__ bmax) {
  const int idx = blockIdx.x * blockDim.x + threadIdx.x;  // b*576 + q
  if (idx >= 8 * 576) return;
  const int b = idx / 576, q = idx % 576;
  const float* base = c + (size_t)b * 576 * 576 + q;
  float m = -1e30f;
  for (int p = 0; p < 576; ++p) m = fmaxf(m, base[(size_t)p * 576]);
  bmax[idx] = m;
}

__global__ void mm_apply_kernel(const float* __restrict__ in,
                                const float* __restrict__ amax,
                                const float* __restrict__ bmax,
                                float* __restrict__ out) {
  const int idx = blockIdx.x * blockDim.x + threadIdx.x;
  if (idx >= 8 * 576 * 576) return;
  const int q = idx % 576;
  const int p = (idx / 576) % 576;
  const int b = idx / (576 * 576);
  const float v = in[idx];
  out[idx] = v * (v / (amax[b * 576 + p] + EPS_MM))
               * (v / (bmax[b * 576 + q] + EPS_MM));
}

// ---------------------------------------------------------------------------
// Conv4d (3x3x3x3, SAME) + ReLU, direct.
// grid = (24*24 over (d1,d2), Cout, B), block = (24,24) over (d3,d4)
// weights for the block's output channel staged in LDS.
// ---------------------------------------------------------------------------
template <int CIN>
__global__ __launch_bounds__(576) void conv4d_relu_kernel(
    const float* __restrict__ x, const float* __restrict__ wgt,
    const float* __restrict__ bias, float* __restrict__ y) {
  __shared__ float ws_[CIN * 81];
  const int co = blockIdx.y;
  const int b  = blockIdx.z;
  const int i1 = blockIdx.x / 24, i2 = blockIdx.x % 24;
  const int i3 = threadIdx.y,     i4 = threadIdx.x;
  const int tid = threadIdx.y * 24 + threadIdx.x;

  for (int i = tid; i < CIN * 81; i += 576) ws_[i] = wgt[co * CIN * 81 + i];
  __syncthreads();

  float acc = bias[co];
  for (int ci = 0; ci < CIN; ++ci) {
#pragma unroll
    for (int k1 = 0; k1 < 3; ++k1) {
      const int j1 = i1 + k1 - 1;
      if ((unsigned)j1 >= 24u) continue;
#pragma unroll
      for (int k2 = 0; k2 < 3; ++k2) {
        const int j2 = i2 + k2 - 1;
        if ((unsigned)j2 >= 24u) continue;
        const float* plane =
            x + (((size_t)b * CIN + ci) * 576 + j1 * 24 + j2) * 576;
        const float* wk = ws_ + ((ci * 3 + k1) * 3 + k2) * 9;
#pragma unroll
        for (int k3 = 0; k3 < 3; ++k3) {
          const int j3 = i3 + k3 - 1;
          if ((unsigned)j3 >= 24u) continue;
#pragma unroll
          for (int k4 = 0; k4 < 3; ++k4) {
            const int j4 = i4 + k4 - 1;
            if ((unsigned)j4 >= 24u) continue;
            acc += plane[j3 * 24 + j4] * wk[k3 * 3 + k4];
          }
        }
      }
    }
  }
  y[(((size_t)b * gridDim.y + co) * 576 + blockIdx.x) * 576 + i3 * 24 + i4] =
      fmaxf(acc, 0.f);
}

// ---------------------------------------------------------------------------
// Launcher
// ---------------------------------------------------------------------------
extern "C" void kernel_launch(void* const* d_in, const int* in_sizes, int n_in,
                              void* d_out, int out_size, void* d_ws, size_t ws_size,
                              hipStream_t stream) {
  const float* featA = (const float*)d_in[0];
  const float* featB = (const float*)d_in[1];
  const float* w1 = (const float*)d_in[2];
  const float* b1 = (const float*)d_in[3];
  const float* w2 = (const float*)d_in[4];
  const float* b2 = (const float*)d_in[5];
  const float* w3 = (const float*)d_in[6];
  const float* b3 = (const float*)d_in[7];
  float* out = (float*)d_out;

  // Workspace layout (all 16B aligned). corrFull region is reused for the
  // conv intermediates once pooling has consumed the full correlation.
  const size_t szCorr   = (size_t)8 * 2304 * 2304;   // floats (169.9 MB)
  const size_t szPooled = (size_t)8 * 576 * 576;     // floats (10.6 MB)
  const size_t szMax    = (size_t)8 * 576;           // floats
  const size_t szX2     = (size_t)8 * 10 * 576 * 576;// floats (106.2 MB)
  const size_t szFeat   = (size_t)8 * 2304 * 1024;   // halves (37.7 MB each)

  char* base = (char*)d_ws;
  float*     corrFull = (float*)base;
  float*     pooled   = (float*)(base + szCorr * 4);
  float*     amax     = pooled + szPooled;
  float*     bmax     = amax + szMax;
  float*     x2       = bmax + szMax;
  _Float16*  faH      = (_Float16*)(x2 + szX2);
  _Float16*  fbH      = faH + szFeat;
  float*     x1 = corrFull;   // 106 MB fits in the 170 MB corr region
  float*     x3 = corrFull;   // x1 dead when conv3 writes (conv3 reads x2)

  // 1) L2-norm + f16 transpose
  {
    dim3 blk(48, 8), grd(48, 8);
    l2norm_f16_kernel<<<grd, blk, 0, stream>>>(featA, faH);
    l2norm_f16_kernel<<<grd, blk, 0, stream>>>(featB, fbH);
  }

  // 2) Correlation GEMM (WMMA, async LDS double-buffer) + fused ReLU + norm
  {
    dim3 grd(2304 / 128, 2304 / 128, 8);
    corr_wmma_kernel<<<grd, 256, 0, stream>>>(faH, fbH, corrFull);
  }

  // 3) 4D max pool k=2
  {
    const int n = 8 * 576 * 576;
    pool4d_kernel<<<(n + 255) / 256, 256, 0, stream>>>(corrFull, pooled);
  }

  // 4) mutual matching #1 (in place on pooled)
  {
    rowmax_kernel<<<8 * 576, 64, 0, stream>>>(pooled, amax);
    colmax_kernel<<<(8 * 576 + 255) / 256, 256, 0, stream>>>(pooled, bmax);
    const int n = 8 * 576 * 576;
    mm_apply_kernel<<<(n + 255) / 256, 256, 0, stream>>>(pooled, amax, bmax, pooled);
  }

  // 5) neighbourhood consensus: 3x Conv4d + ReLU
  {
    dim3 blk(24, 24);
    conv4d_relu_kernel<1><<<dim3(576, 10, 8), blk, 0, stream>>>(pooled, w1, b1, x1);
    conv4d_relu_kernel<10><<<dim3(576, 10, 8), blk, 0, stream>>>(x1, w2, b2, x2);
    conv4d_relu_kernel<10><<<dim3(576, 1, 8), blk, 0, stream>>>(x2, w3, b3, x3);
  }

  // 6) mutual matching #2 -> d_out
  {
    rowmax_kernel<<<8 * 576, 64, 0, stream>>>(x3, amax);
    colmax_kernel<<<(8 * 576 + 255) / 256, 256, 0, stream>>>(x3, bmax);
    const int n = 8 * 576 * 576;
    mm_apply_kernel<<<(n + 255) / 256, 256, 0, stream>>>(x3, amax, bmax, out);
  }
}
